// BipartiteGCNN_82394652607046
// MI455X (gfx1250) — matrix operand order repaired and checked by reading, not compiled, not run on hardware
//
#include <hip/hip_runtime.h>
#include <hip/hip_fp16.h>

typedef __attribute__((ext_vector_type(16))) _Float16 v16h;
typedef __attribute__((ext_vector_type(8)))  _Float16 v8h;
typedef __attribute__((ext_vector_type(8)))  float    v8f;

#define H_DIM 64

__device__ __forceinline__ v8f wmma16(v16h a, v16h b, v8f c) {
  // D = A(16x32 f16) * B(32x16 f16) + C(16x16 f32)
  return __builtin_amdgcn_wmma_f32_16x16x32_f16(false, a, false, b, (short)0, c,
                                                false, false);
}

__device__ __forceinline__ v16h combine16(v8h lo, v8h hi) {
  return __builtin_shufflevector(lo, hi, 0, 1, 2, 3, 4, 5, 6, 7,
                                 8, 9, 10, 11, 12, 13, 14, 15);
}

// ---------------------------------------------------------------------------
// Repack f32 weights into f16 WMMA-B fragment-packed layout.
// Fragment layout (B 32x16 f16): lane l holds column n = l&15,
// k = (l>>4)*8 + j for j<8, 16 + (l>>4)*8 + (j-8) for j>=8, value = W[K0 + k][N0 + n].
// Packed: pw[frag*512 + l*16 + j]  (32 contiguous bytes per lane per frag).
// 48 fragments total: [v2c_W1 rows0-63 | v2c_W1 rows65-128 | v2c_W2 |
//                      c2v_W1 rows0-63 | c2v_W1 rows65-128 | c2v_W2], each
// group = 8 frags ordered [kc][nt].
// ---------------------------------------------------------------------------
__global__ void repack_w_kernel(const float* __restrict__ w1a,
                                const float* __restrict__ w2a,
                                const float* __restrict__ w1b,
                                const float* __restrict__ w2b,
                                _Float16* __restrict__ pw) {
  const int b = blockIdx.x;    // 0..47
  const int t = threadIdx.x;   // 0..511
  const int grp = b >> 3;
  const int g = b & 7;
  const int kc = g >> 2;
  const int nt = g & 3;
  const float* src;
  int rowBase;
  switch (grp) {
    case 0:  src = w1a; rowBase = 0;  break;
    case 1:  src = w1a; rowBase = 65; break;
    case 2:  src = w2a; rowBase = 0;  break;
    case 3:  src = w1b; rowBase = 0;  break;
    case 4:  src = w1b; rowBase = 65; break;
    default: src = w2b; rowBase = 0;  break;
  }
  const int l = t >> 4;   // lane 0..31
  const int j = t & 15;
  const int row = rowBase + kc * 32 + (j < 8 ? (l >> 4) * 8 + j : 16 + (l >> 4) * 8 + (j - 8));
  const int col = nt * 16 + (l & 15);
  pw[b * 512 + t] = (_Float16)src[row * H_DIM + col];
}

// ---------------------------------------------------------------------------
// emb = relu(feat @ W + b); write f16 copy (for WMMA gathers) and f32 copy
// that serves as the residual base / scatter accumulator.
// ---------------------------------------------------------------------------
__global__ void embed_kernel(const float* __restrict__ feat,
                             const float* __restrict__ W,
                             const float* __restrict__ bias, int nNodes,
                             int inDim, _Float16* __restrict__ emb16,
                             float* __restrict__ accumInit) {
  const int idx = blockIdx.x * blockDim.x + threadIdx.x;
  if (idx >= nNodes * H_DIM) return;
  const int node = idx >> 6;
  const int h = idx & 63;
  float acc = bias[h];
  const float* f = feat + (long)node * inDim;
  for (int k = 0; k < inDim; ++k) acc = fmaf(f[k], W[k * H_DIM + h], acc);
  acc = fmaxf(acc, 0.f);
  emb16[idx] = (_Float16)acc;
  accumInit[idx] = acc;
}

__global__ void f32_to_f16_kernel(const float* __restrict__ src,
                                  _Float16* __restrict__ dst, int n) {
  const int i = blockIdx.x * blockDim.x + threadIdx.x;
  if (i < n) dst[i] = (_Float16)src[i];
}

__global__ void out_kernel(const float* __restrict__ nodeEmb,
                           const float* __restrict__ outW,
                           const float* __restrict__ outB,
                           float* __restrict__ scores, int n) {
  const int c = blockIdx.x * blockDim.x + threadIdx.x;
  if (c >= n) return;
  float acc = outB[0];
  const float* p = nodeEmb + (long)c * H_DIM;
  for (int k = 0; k < H_DIM; ++k) acc = fmaf(p[k], outW[k], acc);
  scores[c] = acc;
}

// ---------------------------------------------------------------------------
// Edge MLP + scatter.  Input per edge e: x = [embA[idxA[e]](64) | ef[e](1) |
// embB[idxB[e]](64)].  msg = relu(x@W1+b1)@W2+b2.  accum[idxB[e]] += msg.
// One wave handles a 16-edge tile; 64 outputs via 4 N-tiles of WMMA.
// Layer-1 K: 2 chunks from embA + 2 chunks from embB; edge-feature column is
// the rank-1 term ef*W1[64][:] folded into the accumulator init.
// All 24 weight fragments are staged in LDS once per block and loaded
// just-in-time via ds_load_b128 pairs, keeping VGPR pressure low so the
// random embedding gathers can stay in flight and occupancy stays high.
// ---------------------------------------------------------------------------
__global__ __launch_bounds__(256, 1) void edge_mlp_scatter(
    const _Float16* __restrict__ embA, const _Float16* __restrict__ embB,
    const int* __restrict__ idxA, const int* __restrict__ idxB,
    const float* __restrict__ efeat, const _Float16* __restrict__ pw,
    const float* __restrict__ w1e,  // W1 row 64 (edge-feature row), 64 floats
    const float* __restrict__ b1, const float* __restrict__ b2,
    float* __restrict__ accum, int E) {
  __shared__ __align__(16) _Float16 ldsW[24 * 512];      // 24 KB weight frags
  __shared__ __align__(16) _Float16 ldsH1[8][16 * H_DIM]; // 16 KB relu tiles
  const int lane = threadIdx.x & 31;
  const int wib = threadIdx.x >> 5;
  const int n = lane & 15;    // output column within N-tile / A row index
  const int hl = lane >> 4;   // half-wave select
  _Float16* myLds = &ldsH1[wib][0];

  // Stage the packed weight fragments into LDS (once per block).
  for (int i = threadIdx.x; i < 24 * 512 / 8; i += blockDim.x)
    *(v8h*)(ldsW + i * 8) = *(const v8h*)(pw + i * 8);
  __syncthreads();

  const _Float16* wfrag = ldsW + lane * 16;  // per-lane fragment base

  float b1n[4], b2n[4], w1en[4];
#pragma unroll
  for (int nt = 0; nt < 4; ++nt) {
    b1n[nt] = b1[nt * 16 + n];
    b2n[nt] = b2[nt * 16 + n];
    w1en[nt] = w1e[nt * 16 + n];
  }

  const int ntiles = (E + 15) >> 4;
  const int gwave = blockIdx.x * (blockDim.x >> 5) + wib;
  const int nwave = gridDim.x * (blockDim.x >> 5);

  for (int t = gwave; t < ntiles; t += nwave) {
    const int e0 = t << 4;
    int em = e0 + n;
    if (em >= E) em = E - 1;
    const long ia = idxA[em];
    const long ib = idxB[em];
    const _Float16* pa = embA + ia * H_DIM;
    const _Float16* pb = embB + ib * H_DIM;

    // A fragments (16-bit A 16x32 layout): lane holds row m=n, runs
    // {K0+hl*8 .. +7} and {K0+16+hl*8 .. +7} -> two b128 loads each.
    v16h A1a[2], A1b[2];
#pragma unroll
    for (int kc = 0; kc < 2; ++kc) {
      A1a[kc] = combine16(*(const v8h*)(pa + kc * 32 + hl * 8),
                          *(const v8h*)(pa + kc * 32 + 16 + hl * 8));
      A1b[kc] = combine16(*(const v8h*)(pb + kc * 32 + hl * 8),
                          *(const v8h*)(pb + kc * 32 + 16 + hl * 8));
    }

    // Per-accumulator-row edge features (C/D row = v + 8*hl).
    float ev[8];
#pragma unroll
    for (int v = 0; v < 8; ++v) {
      int ee = e0 + v + 8 * hl;
      ev[v] = efeat[ee < E ? ee : 0];
    }

    // ---- layer 1 + relu -> LDS (f16, row-major 16x64 per-wave tile) ----
#pragma unroll
    for (int nt = 0; nt < 4; ++nt) {
      v8f c;
#pragma unroll
      for (int v = 0; v < 8; ++v) c[v] = fmaf(ev[v], w1en[nt], b1n[nt]);
#pragma unroll
      for (int kc = 0; kc < 2; ++kc) {
        v16h bA = combine16(*(const v8h*)(wfrag + (kc * 4 + nt) * 512),
                            *(const v8h*)(wfrag + (kc * 4 + nt) * 512 + 8));
        c = wmma16(A1a[kc], bA, c);
      }
#pragma unroll
      for (int kc = 0; kc < 2; ++kc) {
        v16h bB = combine16(*(const v8h*)(wfrag + (8 + kc * 4 + nt) * 512),
                            *(const v8h*)(wfrag + (8 + kc * 4 + nt) * 512 + 8));
        c = wmma16(A1b[kc], bB, c);
      }
#pragma unroll
      for (int v = 0; v < 8; ++v) {
        float r = fmaxf(c[v], 0.f);
        myLds[(v + 8 * hl) * H_DIM + nt * 16 + n] = (_Float16)r;
      }
    }

    // cross-lane D->A relayout through per-wave LDS tile
    asm volatile("s_wait_dscnt 0" ::: "memory");
    __builtin_amdgcn_wave_barrier();

    v16h A2[2];
#pragma unroll
    for (int kc = 0; kc < 2; ++kc)
      A2[kc] =
          combine16(*(const v8h*)(myLds + n * H_DIM + kc * 32 + hl * 8),
                    *(const v8h*)(myLds + n * H_DIM + kc * 32 + 16 + hl * 8));

    // Scatter indices, loaded late so they aren't live across layer 1.
    int sidx[8];
#pragma unroll
    for (int v = 0; v < 8; ++v) {
      int ee = e0 + v + 8 * hl;
      sidx[v] = idxB[ee < E ? ee : 0];
    }

    // ---- layer 2 + atomic scatter ----
#pragma unroll
    for (int nt = 0; nt < 4; ++nt) {
      v8f d;
#pragma unroll
      for (int v = 0; v < 8; ++v) d[v] = b2n[nt];
#pragma unroll
      for (int kc = 0; kc < 2; ++kc) {
        v16h bW = combine16(*(const v8h*)(wfrag + (16 + kc * 4 + nt) * 512),
                            *(const v8h*)(wfrag + (16 + kc * 4 + nt) * 512 + 8));
        d = wmma16(A2[kc], bW, d);
      }
#pragma unroll
      for (int v = 0; v < 8; ++v) {
        if (e0 + v + 8 * hl < E)
          unsafeAtomicAdd(&accum[(long)sidx[v] * H_DIM + nt * 16 + n], d[v]);
      }
    }
  }
}

// ---------------------------------------------------------------------------
extern "C" void kernel_launch(void* const* d_in, const int* in_sizes, int n_in,
                              void* d_out, int out_size, void* d_ws,
                              size_t ws_size, hipStream_t stream) {
  const float* colF = (const float*)d_in[0];
  const float* rowF = (const float*)d_in[1];
  const float* eF = (const float*)d_in[2];
  const int* eIdx = (const int*)d_in[3];  // [2, E]: row idx then col idx
  const float* colW = (const float*)d_in[4];
  const float* colB = (const float*)d_in[5];
  const float* rowW = (const float*)d_in[6];
  const float* rowB = (const float*)d_in[7];
  const float* v2cW1 = (const float*)d_in[8];
  const float* v2cB1 = (const float*)d_in[9];
  const float* v2cW2 = (const float*)d_in[10];
  const float* v2cB2 = (const float*)d_in[11];
  const float* c2vW1 = (const float*)d_in[12];
  const float* c2vB1 = (const float*)d_in[13];
  const float* c2vW2 = (const float*)d_in[14];
  const float* c2vB2 = (const float*)d_in[15];
  const float* outW = (const float*)d_in[16];
  const float* outB = (const float*)d_in[17];
  float* scores = (float*)d_out;

  const int nCols = in_sizes[0] / 19;
  const int nRows = in_sizes[1] / 14;
  const int E = in_sizes[2];

  char* wsp = (char*)d_ws;
  auto wsAlloc = [&](size_t bytes) -> void* {
    void* p = (void*)wsp;
    wsp += (bytes + 255) & ~(size_t)255;
    return p;
  };
  float* newRow = (float*)wsAlloc((size_t)nRows * H_DIM * 4);
  float* newCol = (float*)wsAlloc((size_t)nCols * H_DIM * 4);
  _Float16* colH = (_Float16*)wsAlloc((size_t)nCols * H_DIM * 2);
  _Float16* rowH = (_Float16*)wsAlloc((size_t)nRows * H_DIM * 2);
  _Float16* newRowH = (_Float16*)wsAlloc((size_t)nRows * H_DIM * 2);
  _Float16* pw = (_Float16*)wsAlloc(48 * 512 * 2);
  (void)ws_size;
  (void)n_in;
  (void)out_size;

  // 1) pack weights into WMMA-B fragment layout (f16)
  repack_w_kernel<<<48, 512, 0, stream>>>(v2cW1, v2cW2, c2vW1, c2vW2, pw);

  // 2) node embeddings (also seed the scatter accumulators = residual)
  embed_kernel<<<(nCols * H_DIM + 255) / 256, 256, 0, stream>>>(
      colF, colW, colB, nCols, 19, colH, newCol);
  embed_kernel<<<(nRows * H_DIM + 255) / 256, 256, 0, stream>>>(
      rowF, rowW, rowB, nRows, 14, rowH, newRow);

  const int ntiles = (E + 15) >> 4;
  int blocks = (ntiles + 7) / 8;
  if (blocks > 2048) blocks = 2048;
  if (blocks < 1) blocks = 1;

  // 3) v2c:  x = [col[ci] | ef | row[ri]],  newRow[ri] += msg
  edge_mlp_scatter<<<blocks, 256, 0, stream>>>(
      colH, rowH, eIdx + E, eIdx, eF, pw, v2cW1 + 64 * H_DIM, v2cB1, v2cB2,
      newRow, E);

  // 4) convert updated row embeddings to f16 for the next gather phase
  f32_to_f16_kernel<<<(nRows * H_DIM + 255) / 256, 256, 0, stream>>>(
      newRow, newRowH, nRows * H_DIM);

  // 5) c2v:  x = [newRow[ri] | ef | col[ci]],  newCol[ci] += msg
  edge_mlp_scatter<<<blocks, 256, 0, stream>>>(
      newRowH, colH, eIdx, eIdx + E, eF, pw + 24 * 512, c2vW1 + 64 * H_DIM,
      c2vB1, c2vB2, newCol, E);

  // 6) output head
  out_kernel<<<(nCols + 255) / 256, 256, 0, stream>>>(newCol, outW, outB,
                                                      scores, nCols);
}
